// NSAMSA_11759620457190
// MI455X (gfx1250) — matrix operand order repaired and checked by reading, not compiled, not run on hardware
//
#include <hip/hip_runtime.h>
#include <hip/hip_bf16.h>

// ---------------------------------------------------------------------------
// Problem constants (match reference)
// ---------------------------------------------------------------------------
#define DIMX        512
#define NHEADS      8
#define EDIM        64          // DIMX / NHEADS
#define BALL        64
#define TOPK        2
#define NBATCH      2
#define NBALLS      1024
#define NTOTAL      (NBATCH * NBALLS * BALL)   // 131072
#define PERBATCH    (NBALLS * BALL)            // 65536
#define NKEYS       (TOPK * BALL)              // 128
#define KVROWS      (NBATCH * NHEADS * BALL)   // 1024 rows (balls 0..7 per batch)

typedef __attribute__((ext_vector_type(16))) _Float16 v16h;
typedef __attribute__((ext_vector_type(8)))  _Float16 v8h;
typedef __attribute__((ext_vector_type(8)))  float    v8f;
typedef __attribute__((ext_vector_type(4)))  unsigned int u32x4;
typedef __attribute__((ext_vector_type(8)))  int i32x8;
typedef __attribute__((ext_vector_type(4)))  int i32x4;

// ---------------------------------------------------------------------------
// Tensor Data Mover availability (gfx1250). ROCm7.2/clang-22: 5-arg builtin;
// amdgpu-toolchain/clang-23: 6-arg builtin. Fallback: per-lane vector staging.
// ---------------------------------------------------------------------------
#if defined(__AMDGCN__) && defined(__has_builtin)
#  if __has_builtin(__builtin_amdgcn_tensor_load_to_lds) && \
      __has_builtin(__builtin_amdgcn_s_wait_tensorcnt)
#    define HAVE_TDM 1
#  endif
#endif
#ifndef HAVE_TDM
#  define HAVE_TDM 0
#endif

#if HAVE_TDM
// 2-D tile DMA: global (row-major, 2-byte elems, row stride `stride_e` elems)
// -> LDS at byte offset `lds_byte`. tile_w elems x tile_h rows, packed in LDS.
__device__ __forceinline__ void tdm_load_2d(const void* gptr, unsigned lds_byte,
                                            unsigned tensor_w, unsigned tensor_h,
                                            unsigned stride_e,
                                            unsigned tile_w, unsigned tile_h) {
  const unsigned long long ga = (unsigned long long)(uintptr_t)gptr;
  u32x4 g0;
  g0.x = 1u;                                            // count=1, user D#
  g0.y = lds_byte;                                      // lds_addr (bytes)
  g0.z = (unsigned)(ga & 0xffffffffu);                  // global_addr[31:0]
  g0.w = (unsigned)((ga >> 32) & 0x1ffffffu) | (2u << 30); // [56:32] | type=2
  i32x8 g1;
  g1[0] = (int)(1u << 16);                              // wg_mask=0, data_size=2B
  g1[1] = (int)((tensor_w & 0xffffu) << 16);            // tensor_dim0[15:0]
  g1[2] = (int)((tensor_w >> 16) | ((tensor_h & 0xffffu) << 16));
  g1[3] = (int)((tensor_h >> 16) | ((tile_w & 0xffffu) << 16)); // tile_dim0
  g1[4] = (int)(tile_h & 0xffffu);                      // tile_dim1, tile_dim2=0
  g1[5] = (int)stride_e;                                // tensor_dim0_stride lo32
  g1[6] = 0;
  g1[7] = 0;
  const i32x4 gz = {0, 0, 0, 0};
#if __clang_major__ >= 23
  const i32x8 gz8 = {0, 0, 0, 0, 0, 0, 0, 0};
  __builtin_amdgcn_tensor_load_to_lds(g0, g1, gz, gz, gz8, 0);
#else
  __builtin_amdgcn_tensor_load_to_lds(g0, g1, gz, gz, 0);
#endif
}
#endif

// ---------------------------------------------------------------------------
// WMMA helpers (gfx1250 16x16x32 f16 -> f32)
// ---------------------------------------------------------------------------
__device__ __forceinline__ v8f wmma16(v16h a, v16h b, v8f c) {
  return __builtin_amdgcn_wmma_f32_16x16x32_f16(
      /*neg_a=*/false, a, /*neg_b=*/false, b,
      /*c_mod=*/(short)0, c, /*reuse_a=*/false, /*reuse_b=*/false);
}

// A-matrix fragment (16x32, 16-bit): lane L holds row M=L%16; element i maps to
// K = 16*(i>>3) + 8*(L/16) + (i&7).  rowbase points at K=0 of this lane's row.
__device__ __forceinline__ v16h load_a_frag(const _Float16* rowbase, int kh) {
  v8h a = *(const v8h*)(rowbase + 8 * kh);
  v8h b = *(const v8h*)(rowbase + 8 * kh + 16);
  v16h r;
#pragma unroll
  for (int i = 0; i < 8; ++i) { r[i] = a[i]; r[i + 8] = b[i]; }
  return r;
}

// B-matrix fragment (32x16, 16-bit) read from a transposed (NxK) buffer:
// lane L holds column N=L%16; element i maps to K = 16*(i>>3) + 8*(L/16) + (i&7).
__device__ __forceinline__ v16h load_b_frag(const _Float16* rowbase, int kh) {
  v8h a = *(const v8h*)(rowbase + 8 * kh);
  v8h b = *(const v8h*)(rowbase + 8 * kh + 16);
  v16h r;
#pragma unroll
  for (int i = 0; i < 8; ++i) { r[i] = a[i]; r[i + 8] = b[i]; }
  return r;
}

// ---------------------------------------------------------------------------
// Kernel 0: repack weights (f32 -> f16, transposed, QKV de-interleaved)
// ---------------------------------------------------------------------------
__global__ __launch_bounds__(256) void repack_kernel(
    const float* __restrict__ qkv_w, const float* __restrict__ qkv_b,
    const float* __restrict__ proj_w,
    _Float16* __restrict__ WqT, _Float16* __restrict__ WkvT,
    _Float16* __restrict__ WpT, float* __restrict__ bq, float* __restrict__ bkv)
{
  int id = blockIdx.x * 256 + threadIdx.x;
  if (id < 512 * 512) {
    int c = id >> 9, d = id & 511;
    WqT[id] = (_Float16)qkv_w[d * 1536 + 3 * c];
    return;
  }
  id -= 512 * 512;
  if (id < 256 * 512) {
    int c = id >> 9, d = id & 511;
    int col = (c < 128) ? (3 * c + 1) : (3 * (c - 128) + 2);
    WkvT[id] = (_Float16)qkv_w[d * 1536 + col];
    return;
  }
  id -= 256 * 512;
  if (id < 512 * 512) {
    int c = id >> 9, d = id & 511;
    WpT[id] = (_Float16)proj_w[d * 512 + c];
    return;
  }
  id -= 512 * 512;
  if (id < 512) { bq[id] = qkv_b[3 * id]; return; }
  id -= 512;
  if (id < 256) {
    bkv[id] = (id < 128) ? qkv_b[3 * id + 1] : qkv_b[3 * (id - 128) + 2];
  }
}

// ---------------------------------------------------------------------------
// Kernel 1: x' = x + rel @ pe_w + pe_b  (per-ball centered positions), -> f16
// ---------------------------------------------------------------------------
__global__ __launch_bounds__(256) void prep_kernel(
    const float* __restrict__ x, const float* __restrict__ pos,
    const float* __restrict__ pe_w, const float* __restrict__ pe_b,
    _Float16* __restrict__ xb)
{
  __shared__ float sp[BALL][3];
  __shared__ float mean[3];
  const int ball = blockIdx.x;
  const int tid = threadIdx.x;
  if (tid < BALL) {
#pragma unroll
    for (int d = 0; d < 3; ++d) sp[tid][d] = pos[(size_t)(ball * BALL + tid) * 3 + d];
  }
  __syncthreads();
  if (tid < 3) {
    float s = 0.f;
    for (int m2 = 0; m2 < BALL; ++m2) s += sp[m2][tid];
    mean[tid] = s * (1.0f / BALL);
  }
  __syncthreads();
  const size_t base = (size_t)ball * BALL * DIMX;
  for (int i = tid; i < BALL * DIMX; i += 256) {
    int m2 = i >> 9, c = i & 511;
    float r0 = sp[m2][0] - mean[0];
    float r1 = sp[m2][1] - mean[1];
    float r2 = sp[m2][2] - mean[2];
    float v = x[base + i] + r0 * pe_w[c] + r1 * pe_w[512 + c] + r2 * pe_w[1024 + c] + pe_b[c];
    xb[base + i] = (_Float16)v;
  }
}

// ---------------------------------------------------------------------------
// GEMM: C[M][N] = A[M][K=512] * BT[N][K]^T + bias, f16 in / f16 or f32 out.
// 128x128 tile per 256-thread block (8 waves, 4x2 layout, 32x64 per wave).
// TDM double-buffered LDS staging; ROWMAP==1 remaps A tile rows (compact KV).
// ---------------------------------------------------------------------------
template <int ROWMAP, int OUTF32>
__global__ __launch_bounds__(256) void gemm_kernel(
    const _Float16* __restrict__ A, const _Float16* __restrict__ BT,
    const float* __restrict__ bias, void* __restrict__ Cout, int N)
{
  constexpr int K = 512;
  constexpr int NKT = K / 32;                 // 16 K-steps
  __shared__ _Float16 As[2][128 * 32];
  __shared__ _Float16 Bs[2][128 * 32];

  const int tid  = threadIdx.x;
  const int lane = tid & 31;
  const int wave = tid >> 5;
  const int kh   = lane >> 4;      // half-wave selector
  const int l16  = lane & 15;
  const int wm   = wave & 3;       // 4 waves along M
  const int wn   = wave >> 2;      // 2 waves along N
  const int m0   = blockIdx.y * 128;
  const int n0   = blockIdx.x * 128;

  long m0g = m0;
  if (ROWMAP == 1) m0g = (m0 < 512) ? m0 : (PERBATCH + (m0 - 512));

  const v8f vzero = {0.f, 0.f, 0.f, 0.f, 0.f, 0.f, 0.f, 0.f};
  v8f acc[2][4];
#pragma unroll
  for (int mi = 0; mi < 2; ++mi)
#pragma unroll
    for (int ni = 0; ni < 4; ++ni) acc[mi][ni] = vzero;

#if HAVE_TDM
  const unsigned ldsA = (unsigned)(uintptr_t)&As[0][0];
  const unsigned ldsB = (unsigned)(uintptr_t)&Bs[0][0];
  const _Float16* Abase = A + (size_t)m0g * K;
  const _Float16* Bbase = BT + (size_t)n0 * K;
  if (wave == 0) {   // prologue DMA: tiles for k-step 0 into buffer 0
    tdm_load_2d(Abase, ldsA, K, 128, K, 32, 128);
    tdm_load_2d(Bbase, ldsB, K, 128, K, 32, 128);
  }
#else
  const int srow  = tid >> 1;
  const int spart = (tid & 1) * 16;
  long arow_g = m0 + srow;
  if (ROWMAP == 1) arow_g = (arow_g < 512) ? arow_g : (PERBATCH + (arow_g - 512));
  const _Float16* Ag = A + (size_t)arow_g * K + spart;
  const _Float16* Bg = BT + (size_t)(n0 + srow) * K + spart;
#endif

  for (int it = 0; it < NKT; ++it) {
    const int buf = it & 1;
#if HAVE_TDM
    if (wave == 0) {
      if (it + 1 < NKT) {   // DMA next K-tile into the other buffer
        const int ktn = (it + 1) * 32;
        tdm_load_2d(Abase + ktn, ldsA + (buf ^ 1) * 8192u, K, 128, K, 32, 128);
        tdm_load_2d(Bbase + ktn, ldsB + (buf ^ 1) * 8192u, K, 128, K, 32, 128);
        __builtin_amdgcn_s_wait_tensorcnt(2);   // oldest pair (this buf) done
      } else {
        __builtin_amdgcn_s_wait_tensorcnt(0);
      }
    }
    __syncthreads();
#else
    const int kt = it * 32;
    __syncthreads();
    *(v16h*)&As[buf][srow * 32 + spart] = *(const v16h*)(Ag + kt);
    *(v16h*)&Bs[buf][srow * 32 + spart] = *(const v16h*)(Bg + kt);
    if (kt + 32 < K) {
      __builtin_prefetch(Ag + kt + 32, 0, 0);
      __builtin_prefetch(Bg + kt + 32, 0, 0);
    }
    __syncthreads();
#endif

    v16h af[2], bf[4];
#pragma unroll
    for (int mi = 0; mi < 2; ++mi)
      af[mi] = load_a_frag(&As[buf][(wm * 32 + mi * 16 + l16) * 32], kh);
#pragma unroll
    for (int ni = 0; ni < 4; ++ni)
      bf[ni] = load_b_frag(&Bs[buf][(wn * 64 + ni * 16 + l16) * 32], kh);
#pragma unroll
    for (int mi = 0; mi < 2; ++mi)
#pragma unroll
      for (int ni = 0; ni < 4; ++ni)
        acc[mi][ni] = wmma16(af[mi], bf[ni], acc[mi][ni]);

    __syncthreads();   // all waves done with `buf` before it is re-filled
  }

  // Epilogue: D layout: lane l16 = column, VGPR j -> row j + 8*kh.
#pragma unroll
  for (int mi = 0; mi < 2; ++mi) {
#pragma unroll
    for (int ni = 0; ni < 4; ++ni) {
      const int gm = m0 + wm * 32 + mi * 16;
      const int gn = n0 + wn * 64 + ni * 16 + l16;
      const float bv = bias ? bias[gn] : 0.f;
#pragma unroll
      for (int j = 0; j < 8; ++j) {
        const int row = gm + j + 8 * kh;
        const float v = acc[mi][ni][j] + bv;
        if (OUTF32) ((float*)Cout)[(size_t)row * N + gn] = v;
        else        ((_Float16*)Cout)[(size_t)row * N + gn] = (_Float16)v;
      }
    }
  }
}

// ---------------------------------------------------------------------------
// Attention: for (b,h): 128 queries/block attend over 128 keys, E=64.
//   k_sel[j=t*64+m2][e] = KVb[b*512 + h*64 + m2][t*64+e]
//   v_sel[j][e]         = KVb[b*512 + h*64 + m2][128 + t*64+e]
// scores = Q*k_sel^T * 1/8 -> softmax -> out = P*v_sel -> Ob[row][h*64+e]
// ---------------------------------------------------------------------------
__global__ __launch_bounds__(256) void attn_kernel(
    const _Float16* __restrict__ Qb, const _Float16* __restrict__ KVb,
    _Float16* __restrict__ Ob)
{
  __shared__ _Float16 Ks[NKEYS * EDIM];      // [key][e]   (== k_sel, B^T form)
  __shared__ _Float16 Vt[EDIM * NKEYS];      // [e][key]   (== v_sel^T, B^T form)
  __shared__ _Float16 At[8 * 16 * NKEYS];    // wave-private attn probs [q][key]

  const int tid  = threadIdx.x;
  const int lane = tid & 31;
  const int wave = tid >> 5;
  const int kh   = lane >> 4;
  const int l16  = lane & 15;
  const int qb   = blockIdx.x;    // 512 query blocks of 128
  const int h    = blockIdx.y;    // head
  const int bi   = blockIdx.z;    // batch

  const _Float16* kvrow0 = KVb + (size_t)(bi * 512 + h * 64) * 256;

  // ---- stage k_sel: two 64x64 tiles (t = 0,1) ----
#if HAVE_TDM
  if (wave == 0) {
    tdm_load_2d(kvrow0 +   0, (unsigned)(uintptr_t)&Ks[0],       1024, 64, 256, 64, 64);
    tdm_load_2d(kvrow0 +  64, (unsigned)(uintptr_t)&Ks[64 * 64], 1024, 64, 256, 64, 64);
  }
#else
  {
    const int j   = tid >> 1;              // key 0..127
    const int off = (tid & 1) * 32;
    const int t = j >> 6, m2 = j & 63;
    const _Float16* src = kvrow0 + (size_t)m2 * 256 + t * 64 + off;
    _Float16* dst = &Ks[j * 64 + off];
    *(v16h*)dst        = *(const v16h*)src;
    *(v16h*)(dst + 16) = *(const v16h*)(src + 16);
  }
#endif
  // ---- stage v_sel transposed (scalar; TDM cannot transpose) ----
  for (int i = 0; i < 32; ++i) {
    const int idx = tid + i * 256;
    const int e = idx >> 7, j = idx & 127;
    const int t = j >> 6, m2 = j & 63;
    Vt[e * NKEYS + j] = kvrow0[(size_t)m2 * 256 + 128 + t * 64 + e];
  }
#if HAVE_TDM
  if (wave == 0) __builtin_amdgcn_s_wait_tensorcnt(0);
#endif
  __syncthreads();

  // ---- load Q fragments straight from global (A layout matches) ----
  const int q0 = qb * 128 + wave * 16;
  const size_t qrow = (size_t)(bi * PERBATCH + q0 + l16) * DIMX + h * EDIM;
  v16h aq[2];
#pragma unroll
  for (int s = 0; s < 2; ++s) aq[s] = load_a_frag(Qb + qrow + s * 32, kh);

  // ---- scores (16 q x 128 key): batch-load 8 B-frags, then 8 back-to-back
  //      WMMAs so the ds waits amortize over the whole burst ----
  const v8f vzero = {0.f, 0.f, 0.f, 0.f, 0.f, 0.f, 0.f, 0.f};
  v8f sc[8];
#pragma unroll
  for (int n = 0; n < 8; ++n) sc[n] = vzero;
#pragma unroll
  for (int s = 0; s < 2; ++s) {
    v16h kb[8];
#pragma unroll
    for (int n = 0; n < 8; ++n)
      kb[n] = load_b_frag(&Ks[(n * 16 + l16) * 64 + s * 32], kh);
#pragma unroll
    for (int n = 0; n < 8; ++n) sc[n] = wmma16(aq[s], kb[n], sc[n]);
  }

  // ---- softmax over 128 keys per query row (wave32: 16-lane xor reduce) ----
  const float scale = 0.125f;               // 1/sqrt(64)
  const int wbase = wave * 16 * NKEYS;
#pragma unroll
  for (int j = 0; j < 8; ++j) {
    float m = -3.0e38f;
#pragma unroll
    for (int n = 0; n < 8; ++n) m = fmaxf(m, sc[n][j] * scale);
#pragma unroll
    for (int o = 1; o < 16; o <<= 1) m = fmaxf(m, __shfl_xor(m, o, 32));
    float ex[8], sum = 0.f;
#pragma unroll
    for (int n = 0; n < 8; ++n) { ex[n] = __expf(sc[n][j] * scale - m); sum += ex[n]; }
#pragma unroll
    for (int o = 1; o < 16; o <<= 1) sum += __shfl_xor(sum, o, 32);
    const float inv = 1.f / sum;
    const int q = j + 8 * kh;
#pragma unroll
    for (int n = 0; n < 8; ++n)
      At[wbase + q * NKEYS + n * 16 + l16] = (_Float16)(ex[n] * inv);
  }
  // wave-private LDS region: compiler inserts the required s_wait_dscnt

  // ---- out = P (16x128) x v_sel (128x64) ----
  v8f oa[4];
#pragma unroll
  for (int et = 0; et < 4; ++et) oa[et] = vzero;
  const _Float16* ap = &At[wbase + l16 * NKEYS];
#pragma unroll
  for (int s = 0; s < 4; ++s) {
    v16h a = load_a_frag(ap + s * 32, kh);
    v16h vb[4];
#pragma unroll
    for (int et = 0; et < 4; ++et)
      vb[et] = load_b_frag(&Vt[(et * 16 + l16) * NKEYS + s * 32], kh);
#pragma unroll
    for (int et = 0; et < 4; ++et) oa[et] = wmma16(a, vb[et], oa[et]);
  }
#pragma unroll
  for (int et = 0; et < 4; ++et)
#pragma unroll
    for (int j = 0; j < 8; ++j) {
      const size_t row = (size_t)(bi * PERBATCH + q0 + j + 8 * kh);
      Ob[row * DIMX + h * EDIM + et * 16 + l16] = (_Float16)oa[et][j];
    }
}

// ---------------------------------------------------------------------------
// Host-side orchestration
// ---------------------------------------------------------------------------
extern "C" void kernel_launch(void* const* d_in, const int* in_sizes, int n_in,
                              void* d_out, int out_size, void* d_ws, size_t ws_size,
                              hipStream_t stream) {
  (void)in_sizes; (void)n_in; (void)out_size; (void)ws_size;
  const float* x      = (const float*)d_in[0];
  const float* pos    = (const float*)d_in[1];
  const float* qkv_w  = (const float*)d_in[2];
  const float* qkv_b  = (const float*)d_in[3];
  const float* proj_w = (const float*)d_in[4];
  const float* proj_b = (const float*)d_in[5];
  const float* pe_w   = (const float*)d_in[6];
  const float* pe_b   = (const float*)d_in[7];

  char* ws = (char*)d_ws;
  size_t off = 0;
  auto alloc = [&](size_t bytes) -> void* {
    void* p = ws + off;
    off += (bytes + 255) & ~(size_t)255;
    return p;
  };
  _Float16* xb   = (_Float16*)alloc((size_t)NTOTAL * DIMX * 2);   // 128 MB
  _Float16* Qb   = (_Float16*)alloc((size_t)NTOTAL * DIMX * 2);   // 128 MB
  _Float16* Ob   = (_Float16*)alloc((size_t)NTOTAL * DIMX * 2);   // 128 MB
  _Float16* WqT  = (_Float16*)alloc((size_t)512 * 512 * 2);
  _Float16* WkvT = (_Float16*)alloc((size_t)256 * 512 * 2);
  _Float16* WpT  = (_Float16*)alloc((size_t)512 * 512 * 2);
  float*    bq   = (float*)alloc(512 * 4);
  float*    bkv  = (float*)alloc(256 * 4);
  _Float16* KVb  = (_Float16*)alloc((size_t)KVROWS * 256 * 2);

  // 0) repack weights
  {
    const int total = 512 * 512 + 256 * 512 + 512 * 512 + 512 + 256;
    repack_kernel<<<(total + 255) / 256, 256, 0, stream>>>(
        qkv_w, qkv_b, proj_w, WqT, WkvT, WpT, bq, bkv);
  }
  // 1) positional prep -> xb (f16)
  prep_kernel<<<NBATCH * NBALLS, 256, 0, stream>>>(x, pos, pe_w, pe_b, xb);

  // 2) Q = xb @ Wq + bq  (131072 x 512)
  gemm_kernel<0, 0><<<dim3(512 / 128, NTOTAL / 128), 256, 0, stream>>>(
      xb, WqT, bq, (void*)Qb, 512);

  // 3) KV for balls 0..7 of each batch (1024 x 256, tile-remapped A rows)
  gemm_kernel<1, 0><<<dim3(256 / 128, KVROWS / 128), 256, 0, stream>>>(
      xb, WkvT, bkv, (void*)KVb, 256);

  // 4) attention -> Ob (f16)
  attn_kernel<<<dim3(PERBATCH / 128, NHEADS, NBATCH), 256, 0, stream>>>(
      Qb, KVb, Ob);

  // 5) final projection -> d_out (f32)
  gemm_kernel<0, 1><<<dim3(512 / 128, NTOTAL / 128), 256, 0, stream>>>(
      Ob, WpT, proj_b, d_out, 512);
}
